// NonLocal_40707700032186
// MI455X (gfx1250) — matrix operand order repaired and checked by reading, not compiled, not run on hardware
//
#include <hip/hip_runtime.h>

typedef __attribute__((ext_vector_type(16))) __bf16 v16bf;
typedef __attribute__((ext_vector_type(8)))  __bf16 v8bf;
typedef __attribute__((ext_vector_type(8)))  float  v8f;
typedef __attribute__((ext_vector_type(4)))  int    v4i;
typedef __attribute__((address_space(1))) v4i gv4i;  // int4 in global AS
typedef __attribute__((address_space(3))) v4i lv4i;  // int4 in LDS AS

#define B_ 4
#define C_ 128
#define N_ 4096  // 64*64

#if defined(__has_builtin)
#if __has_builtin(__builtin_amdgcn_global_load_async_to_lds_b128) && \
    __has_builtin(__builtin_amdgcn_s_wait_asynccnt)
#define NL_ASYNC_LDS 1
#endif
#endif
#ifndef NL_ASYNC_LDS
#define NL_ASYNC_LDS 0
#endif

__device__ __forceinline__ void async_copy_b128(const __bf16* src, __bf16* dst,
                                                int n16) {
#if NL_ASYNC_LDS
#pragma unroll
  for (int j = 0; j < 8; ++j)
    if (j < n16)
      __builtin_amdgcn_global_load_async_to_lds_b128((gv4i*)(src + 8 * j),
                                                     (lv4i*)(dst + 8 * j), 0, 0);
#else
#pragma unroll
  for (int j = 0; j < 8; ++j)
    if (j < n16) ((uint4*)dst)[j] = ((const uint4*)src)[j];
#endif
}

__device__ __forceinline__ void async_wait() {
#if NL_ASYNC_LDS
  __builtin_amdgcn_s_wait_asynccnt(0);
#endif
}

// ---------------- WMMA fragment helpers (wave32, 16x16x32 bf16) ----------------
// A (16xK=32): lane holds row m=lane%16; K runs {8h+0..7} and {16+8h+0..7} -> 2 vector loads
__device__ __forceinline__ v16bf load_a_bf16(const __bf16* p0, int ld) {
  const int lane = threadIdx.x & 31;
  const int half = lane >> 4, m = lane & 15;
  const __bf16* p = p0 + m * ld + 8 * half;
  const v8bf lo = *(const v8bf*)(p);
  const v8bf hi = *(const v8bf*)(p + 16);
  v16bf a;
#pragma unroll
  for (int j = 0; j < 8; ++j) {
    a[j] = lo[j];
    a[8 + j] = hi[j];
  }
  return a;
}

__device__ __forceinline__ v16bf load_a_f32(const float* p0, int ld) {
  const int lane = threadIdx.x & 31;
  const int half = lane >> 4, m = lane & 15;
  const float* p = p0 + m * ld + 8 * half;
  const float4 f0 = ((const float4*)p)[0];
  const float4 f1 = ((const float4*)p)[1];
  const float4 f2 = ((const float4*)(p + 16))[0];
  const float4 f3 = ((const float4*)(p + 16))[1];
  v16bf a;
  a[0] = (__bf16)f0.x; a[1] = (__bf16)f0.y; a[2] = (__bf16)f0.z; a[3] = (__bf16)f0.w;
  a[4] = (__bf16)f1.x; a[5] = (__bf16)f1.y; a[6] = (__bf16)f1.z; a[7] = (__bf16)f1.w;
  a[8] = (__bf16)f2.x; a[9] = (__bf16)f2.y; a[10] = (__bf16)f2.z; a[11] = (__bf16)f2.w;
  a[12] = (__bf16)f3.x; a[13] = (__bf16)f3.y; a[14] = (__bf16)f3.z; a[15] = (__bf16)f3.w;
  return a;
}

// B (K=32 x 16) from an N-major LDS tile T[n][k] (ld = row stride in elems):
// lane (half,n) needs K = {8*half + 0..7} and {16 + 8*half + 0..7} at column n -> 2x b128
__device__ __forceinline__ v16bf load_b_t(const __bf16* Tn0k0, int ld) {
  const int lane = threadIdx.x & 31;
  const int half = lane >> 4, n = lane & 15;
  const __bf16* p = Tn0k0 + n * ld + 8 * half;
  const v8bf lo = *(const v8bf*)(p);
  const v8bf hi = *(const v8bf*)(p + 16);
  v16bf b;
#pragma unroll
  for (int j = 0; j < 8; ++j) {
    b[j] = lo[j];
    b[8 + j] = hi[j];
  }
  return b;
}

__device__ __forceinline__ v8f wmma_bf16(v16bf a, v16bf b, v8f c) {
  return __builtin_amdgcn_wmma_f32_16x16x32_bf16(false, a, false, b, (short)0, c,
                                                 false, false);
}

// ---------------- Kernel 1: three 1x1 conv projections -> bf16 ----------------
// theta [B][N][C], phi [B][N][C], g [B][C][N]
__global__ __launch_bounds__(256) void nl_proj_kernel(
    const float* __restrict__ x, const float* __restrict__ Wg,
    const float* __restrict__ bg, const float* __restrict__ Wth,
    const float* __restrict__ bth, const float* __restrict__ Wph,
    const float* __restrict__ bph, __bf16* __restrict__ theta,
    __bf16* __restrict__ phi, __bf16* __restrict__ g) {
  __shared__ __align__(16) __bf16 Xt[128][136];  // x tile, N-major: [n][c]
  const int b = blockIdx.x >> 5;
  const int n0 = (blockIdx.x & 31) * 128;
  const int tid = threadIdx.x, wave = tid >> 5, lane = tid & 31;
  const int half = lane >> 4, col = lane & 15;
  const float* xb = x + (size_t)b * C_ * N_;

  {  // fp32 -> bf16 transposed tile; pack channel pairs -> b32 LDS stores
    const int pair = tid >> 2;  // c rows 2*pair, 2*pair+1
    const int quar = tid & 3;   // 32-wide n sub-range
    const int c0 = pair * 2;
    const float4* srcA = (const float4*)(xb + (size_t)c0 * N_ + n0 + quar * 32);
    const float4* srcB =
        (const float4*)(xb + (size_t)(c0 + 1) * N_ + n0 + quar * 32);
#pragma unroll
    for (int j = 0; j < 8; ++j) {
      float4 va = srcA[j], vb = srcB[j];
      const int nn = quar * 32 + 4 * j;
      union { __bf16 h[2]; unsigned u; } p0, p1, p2, p3;
      p0.h[0] = (__bf16)va.x; p0.h[1] = (__bf16)vb.x;
      p1.h[0] = (__bf16)va.y; p1.h[1] = (__bf16)vb.y;
      p2.h[0] = (__bf16)va.z; p2.h[1] = (__bf16)vb.z;
      p3.h[0] = (__bf16)va.w; p3.h[1] = (__bf16)vb.w;
      *(unsigned*)&Xt[nn + 0][c0] = p0.u;
      *(unsigned*)&Xt[nn + 1][c0] = p1.u;
      *(unsigned*)&Xt[nn + 2][c0] = p2.u;
      *(unsigned*)&Xt[nn + 3][c0] = p3.u;
    }
  }
  __syncthreads();

  const float* Ws[3] = {Wg, Wth, Wph};
  const float* bs[3] = {bg, bth, bph};

#pragma unroll
  for (int p = 0; p < 3; ++p) {
    v16bf wa[4];
#pragma unroll
    for (int kt = 0; kt < 4; ++kt)
      wa[kt] = load_a_f32(Ws[p] + (size_t)(wave * 16) * C_ + kt * 32, C_);
    float bias_r[8];
#pragma unroll
    for (int r = 0; r < 8; ++r) bias_r[r] = bs[p][wave * 16 + 8 * half + r];

#pragma unroll
    for (int nt = 0; nt < 8; ++nt) {
      v16bf bf[4];  // preload B-frags so loads clause ahead of the WMMA chain
#pragma unroll
      for (int kt = 0; kt < 4; ++kt)
        bf[kt] = load_b_t(&Xt[nt * 16][kt * 32], 136);
      v8f acc = {};
#pragma unroll
      for (int kt = 0; kt < 4; ++kt) acc = wmma_bf16(wa[kt], bf[kt], acc);
      const int n = n0 + nt * 16 + col;
#pragma unroll
      for (int r = 0; r < 8; ++r) {
        const int c = wave * 16 + 8 * half + r;
        const __bf16 v = (__bf16)(acc[r] + bias_r[r]);
        if (p == 0)
          g[((size_t)b * C_ + c) * N_ + n] = v;      // [B][C][N]
        else if (p == 1)
          theta[((size_t)b * N_ + n) * C_ + c] = v;  // [B][N][C]
        else
          phi[((size_t)b * N_ + n) * C_ + c] = v;    // [B][N][C]
      }
    }
  }
}

// ---------------- Kernel 2: flash non-local attention ----------------
// double-buffered async tiles; each wave owns 16 queries; online softmax
__global__ __launch_bounds__(256) void nl_attn_kernel(
    const __bf16* __restrict__ theta, const __bf16* __restrict__ phi,
    const __bf16* __restrict__ g, __bf16* __restrict__ ybuf) {
  __shared__ __align__(16) __bf16 Pt[2][64][136];   // phi tiles, N-major: [key][c]
  __shared__ __align__(16) __bf16 Gt[2][128][72];   // g tiles,  N-major: [c][key]
  __shared__ __align__(16) __bf16 Ssh[8][16][72];   // per-wave P staging
  const int b = blockIdx.x >> 5;
  const int q0 = (blockIdx.x & 31) * 128;
  const int tid = threadIdx.x, wave = tid >> 5, lane = tid & 31;
  const int half = lane >> 4, col = lane & 15;

  const __bf16* thb = theta + (size_t)b * N_ * C_;  // [q][c]
  const __bf16* phib = phi + (size_t)b * N_ * C_;   // [key][c]
  const __bf16* gb = g + (size_t)b * C_ * N_;       // [c][key]

  const int qw = q0 + wave * 16;
  v16bf qa[4];  // theta A-fragments, resident for whole pass
#pragma unroll
  for (int kt = 0; kt < 4; ++kt)
    qa[kt] = load_a_bf16(thb + (size_t)qw * C_ + kt * 32, C_);

  v8f yacc[8];
  const v8f vzero = {};
#pragma unroll
  for (int i = 0; i < 8; ++i) yacc[i] = vzero;
  float mrow[8], lrow[8];
#pragma unroll
  for (int r = 0; r < 8; ++r) {
    mrow[r] = -3.0e38f;
    lrow[r] = 0.0f;
  }

  auto load_tiles = [&](int buf, int k0) {
    {  // phi tile: 64 rows (key) x 128 (c)
      const int row = tid >> 2, part = tid & 3;
      async_copy_b128(phib + (size_t)(k0 + row) * C_ + part * 32,
                      &Pt[buf][row][part * 32], 4);
    }
    {  // g tile: 128 rows (c) x 64 (key)
      const int row = tid >> 1, part = tid & 1;
      async_copy_b128(gb + (size_t)row * N_ + k0 + part * 32,
                      &Gt[buf][row][part * 32], 4);
    }
  };

  const int NT = N_ / 64;
  load_tiles(0, 0);  // prologue
  async_wait();
  __syncthreads();

  int p = 0;
#pragma unroll 1
  for (int it = 0; it < NT; ++it) {
    const int k0 = it * 64;
    if (it + 1 < NT) load_tiles(p ^ 1, k0 + 64);  // async prefetch, no wait yet
    if (k0 + 128 < N_) __builtin_prefetch(phib + (size_t)(k0 + 128) * C_, 0, 0);

    // logits tile: 16 q x 64 keys, K = C
    v8f s[4];
#pragma unroll
    for (int nt = 0; nt < 4; ++nt) {
      v16bf bf[4];
#pragma unroll
      for (int kc = 0; kc < 4; ++kc)
        bf[kc] = load_b_t(&Pt[p][nt * 16][kc * 32], 136);
      v8f acc = vzero;
#pragma unroll
      for (int kc = 0; kc < 4; ++kc) acc = wmma_bf16(qa[kc], bf[kc], acc);
      s[nt] = acc;
    }

    // online softmax: rows live in lane-halves; xor masks 1..8 stay in-half
    float scale_r[8];
#pragma unroll
    for (int r = 0; r < 8; ++r) {
      float tmax = s[0][r];
#pragma unroll
      for (int nt = 1; nt < 4; ++nt) tmax = fmaxf(tmax, s[nt][r]);
#pragma unroll
      for (int off = 1; off < 16; off <<= 1)
        tmax = fmaxf(tmax, __shfl_xor(tmax, off, 32));
      const float mnew = fmaxf(mrow[r], tmax);
      const float sc = __expf(mrow[r] - mnew);
      float psum = 0.0f;
#pragma unroll
      for (int nt = 0; nt < 4; ++nt) {
        const float pv = __expf(s[nt][r] - mnew);
        s[nt][r] = pv;
        psum += pv;
      }
#pragma unroll
      for (int off = 1; off < 16; off <<= 1) psum += __shfl_xor(psum, off, 32);
      lrow[r] = lrow[r] * sc + psum;
      mrow[r] = mnew;
      scale_r[r] = sc;
    }
#pragma unroll
    for (int ct = 0; ct < 8; ++ct)
#pragma unroll
      for (int r = 0; r < 8; ++r) yacc[ct][r] *= scale_r[r];

    // transpose P from D-layout to A-layout via per-wave LDS staging
#pragma unroll
    for (int nt = 0; nt < 4; ++nt)
#pragma unroll
      for (int r = 0; r < 8; ++r)
        Ssh[wave][8 * half + r][nt * 16 + col] = (__bf16)s[nt][r];
    asm volatile("s_wait_dscnt 0" ::: "memory");
    v16bf pa[2];
#pragma unroll
    for (int kt = 0; kt < 2; ++kt)
      pa[kt] = load_a_bf16(&Ssh[wave][0][0] + kt * 32, 72);

    // y(16 x C) += P(16 x 64) @ g(64 x C)
#pragma unroll
    for (int ct = 0; ct < 8; ++ct) {
      v16bf bf[2];
#pragma unroll
      for (int kt = 0; kt < 2; ++kt)
        bf[kt] = load_b_t(&Gt[p][ct * 16][kt * 32], 72);
#pragma unroll
      for (int kt = 0; kt < 2; ++kt)
        yacc[ct] = wmma_bf16(pa[kt], bf[kt], yacc[ct]);
    }

    async_wait();     // next buffer fully in LDS (this wave's issues)
    __syncthreads();  // everyone done reading buffer p + all async complete
    p ^= 1;
  }

  // normalize and store y as [B][N][C]
  __bf16* yb = ybuf + (size_t)b * N_ * C_;
#pragma unroll
  for (int r = 0; r < 8; ++r) {
    const float inv = 1.0f / lrow[r];
    const int q = qw + 8 * half + r;
#pragma unroll
    for (int ct = 0; ct < 8; ++ct) {
      const int c = ct * 16 + col;
      yb[(size_t)q * C_ + c] = (__bf16)(yacc[ct][r] * inv);
    }
  }
}

// ---------------- Kernel 3: Wo GEMM + bias + sigmoid gate ----------------
__global__ __launch_bounds__(256) void nl_out_kernel(
    const float* __restrict__ x, const float* __restrict__ Wo,
    const float* __restrict__ bo, const __bf16* __restrict__ ybuf,
    float* __restrict__ out) {
  __shared__ __align__(16) __bf16 Yt[128][136];  // y tile, N-major: [n][c]
  const int b = blockIdx.x >> 5;
  const int n0 = (blockIdx.x & 31) * 128;
  const int tid = threadIdx.x, wave = tid >> 5, lane = tid & 31;
  const int half = lane >> 4, col = lane & 15;
  const __bf16* yb = ybuf + (size_t)b * N_ * C_;  // [n][c]

  {  // y tile: 128 rows (n) x 128 (c); contiguous row copies via async-to-LDS
    const int row = tid >> 1, part = tid & 1;
    async_copy_b128(yb + (size_t)(n0 + row) * C_ + part * 64, &Yt[row][part * 64],
                    8);
    async_wait();
  }
  __syncthreads();

  v16bf wa[4];
#pragma unroll
  for (int kt = 0; kt < 4; ++kt)
    wa[kt] = load_a_f32(Wo + (size_t)(wave * 16) * C_ + kt * 32, C_);
  float bias_r[8];
#pragma unroll
  for (int r = 0; r < 8; ++r) bias_r[r] = bo[wave * 16 + 8 * half + r];

  const float* xb = x + (size_t)b * C_ * N_;
  float* ob = out + (size_t)b * C_ * N_;

#pragma unroll
  for (int nt = 0; nt < 8; ++nt) {
    v16bf bf[4];
#pragma unroll
    for (int kt = 0; kt < 4; ++kt) bf[kt] = load_b_t(&Yt[nt * 16][kt * 32], 136);
    v8f acc = {};
#pragma unroll
    for (int kt = 0; kt < 4; ++kt) acc = wmma_bf16(wa[kt], bf[kt], acc);
    const int n = n0 + nt * 16 + col;
#pragma unroll
    for (int r = 0; r < 8; ++r) {
      const int c = wave * 16 + 8 * half + r;
      const float v = acc[r] + bias_r[r];
      const float sg = 1.0f / (1.0f + __expf(-v));
      ob[(size_t)c * N_ + n] = sg * xb[(size_t)c * N_ + n];
    }
  }
}

// ---------------- launcher ----------------
extern "C" void kernel_launch(void* const* d_in, const int* in_sizes, int n_in,
                              void* d_out, int out_size, void* d_ws, size_t ws_size,
                              hipStream_t stream) {
  const float* x = (const float*)d_in[0];
  const float* Wg = (const float*)d_in[1];
  const float* bg = (const float*)d_in[2];
  const float* Wth = (const float*)d_in[3];
  const float* bth = (const float*)d_in[4];
  const float* Wph = (const float*)d_in[5];
  const float* bph = (const float*)d_in[6];
  const float* Wo = (const float*)d_in[7];
  const float* bo = (const float*)d_in[8];
  float* out = (float*)d_out;

  // workspace: 4 bf16 buffers of B*C*N = 4 MiB each
  char* ws = (char*)d_ws;
  const size_t seg = (size_t)B_ * C_ * N_ * sizeof(__bf16);
  __bf16* theta = (__bf16*)(ws + 0 * seg);  // [B][N][C]
  __bf16* phi = (__bf16*)(ws + 1 * seg);    // [B][N][C]
  __bf16* gbuf = (__bf16*)(ws + 2 * seg);   // [B][C][N]
  __bf16* ybuf = (__bf16*)(ws + 3 * seg);   // [B][N][C]

  dim3 grid(B_ * (N_ / 128));
  dim3 block(256);
  nl_proj_kernel<<<grid, block, 0, stream>>>(x, Wg, bg, Wth, bth, Wph, bph, theta,
                                             phi, gbuf);
  nl_attn_kernel<<<grid, block, 0, stream>>>(theta, phi, gbuf, ybuf);
  nl_out_kernel<<<grid, block, 0, stream>>>(x, Wo, bo, ybuf, out);
}
